// CrossAttention_20289425507155
// MI455X (gfx1250) — matrix operand order repaired and checked by reading, not compiled, not run on hardware
//
#include <hip/hip_runtime.h>
#include <hip/hip_bf16.h>

// ---------------------------------------------------------------------------
// CrossAttention for MI455X (gfx1250, wave32, WMMA 16x16x32 f16 -> f32 acc)
//   B=2, Tq=Tkv=2048, D_MODEL=1024, H=16, Hkv=4, HEAD_DIM=64
// Pipeline: f32->f16 convert, WMMA NT-GEMM projections (Q,K,V) with
// double-buffered GLOBAL_LOAD_ASYNC_TO_LDS staging, flash-attention with
// online softmax (async K/Q staging), WMMA NT-GEMM output proj.
// ---------------------------------------------------------------------------

typedef __attribute__((ext_vector_type(16))) _Float16 v16h;
typedef __attribute__((ext_vector_type(8)))  _Float16 v8h;
typedef __attribute__((ext_vector_type(8)))  float    v8f;

#define D_MODEL   1024
#define N_HEADS   16
#define N_KV_HEADS 4
#define HEAD_DIM  64
#define BATCH     2
#define TQ        2048
#define TKV       2048
#define KV_DIM    (N_KV_HEADS * HEAD_DIM)   // 256

// ---------------------------------------------------------------------------
// CDNA5 async global->LDS copy (ASYNCcnt-tracked, no VGPR staging).
// VDST = 32-bit LDS byte address, VADDR = 64-bit global address, no SADDR.
// Generic LDS pointers on gfx1250 carry the LDS offset in the low 32 bits.
// ---------------------------------------------------------------------------
__device__ __forceinline__ void async_ld_b128(_Float16* lds, const _Float16* g) {
  unsigned l = (unsigned)(uintptr_t)lds;
  unsigned long long a = (unsigned long long)(uintptr_t)g;
  asm volatile("global_load_async_to_lds_b128 %0, %1, off"
               :: "v"(l), "v"(a) : "memory");
}
__device__ __forceinline__ void wait_async_le4() {
  asm volatile("s_wait_asynccnt 0x4" ::: "memory");
}
__device__ __forceinline__ void wait_async_0() {
  asm volatile("s_wait_asynccnt 0x0" ::: "memory");
}

// ---------------------------------------------------------------------------
// Fragment loaders (wave32 WMMA layouts per CDNA5 ISA 7.12.2)
// A (16x32 f16, MxK): lane L<16: row M=L, K={0..7,16..23}; lane L+16: K={8..15,24..31}
// B (32x16 f16, KxN): lane L<16: col N=L, K={0..7,16..23}; lane L+16: K={8..15,24..31}
// C/D (16x16 f32):    VGPR v: M = v + 8*(lane>=16), N = lane&15.
// ---------------------------------------------------------------------------
__device__ __forceinline__ v16h load_frag_a(const _Float16* base, int stride, int lane) {
  const int r  = lane & 15;
  const int kb = (lane >> 4) << 3;             // 0 or 8
  const _Float16* row = base + r * stride;
  v8h lo = *(const v8h*)(row + kb);
  v8h hi = *(const v8h*)(row + kb + 16);
  v16h a;
#pragma unroll
  for (int i = 0; i < 8; ++i) { a[i] = lo[i]; a[8 + i] = hi[i]; }
  return a;
}

__device__ __forceinline__ v16h load_frag_b(const _Float16* base, int stride, int lane) {
  const int n  = lane & 15;
  const int ko = (lane >> 4) << 3;             // 0 or 8
  const _Float16* row = base + n * stride + ko;
  v8h lo = *(const v8h*)(row);
  v8h hi = *(const v8h*)(row + 16);
  v16h b;
#pragma unroll
  for (int i = 0; i < 8; ++i) { b[i] = lo[i]; b[8 + i] = hi[i]; }
  return b;
}

#define WMMA_F16(A, B, C) \
  __builtin_amdgcn_wmma_f32_16x16x32_f16(false, (A), false, (B), (short)0, (C), false, false)

// ---------------------------------------------------------------------------
// f32 -> f16 conversion (grid-stride)
// ---------------------------------------------------------------------------
__global__ void cvt_f32_to_f16(const float* __restrict__ in,
                               _Float16* __restrict__ out, size_t n) {
  size_t i = (size_t)blockIdx.x * blockDim.x + threadIdx.x;
  size_t s = (size_t)gridDim.x * blockDim.x;
  for (; i < n; i += s) out[i] = (_Float16)in[i];
}

// ---------------------------------------------------------------------------
// NT GEMM: C[M,N] = A[M,K] * B[N,K]^T   (A,B f16 row-major, K inner)
// Block: 256 threads (8 waves), 128x128 tile; wave = 32x64 (2x4 WMMA tiles).
// K staged through double-buffered LDS in 32-wide slabs via async-to-LDS.
// ---------------------------------------------------------------------------
template <typename OutT>
__global__ __launch_bounds__(256)
void gemm_nt_wmma(const _Float16* __restrict__ A, const _Float16* __restrict__ Bm,
                  OutT* __restrict__ C, int M, int N, int K) {
  __shared__ __align__(16) _Float16 As[2][128 * 32];
  __shared__ __align__(16) _Float16 Bs[2][128 * 32];

  const int tid  = threadIdx.x;
  const int lane = tid & 31;
  const int wave = tid >> 5;
  const int bm = blockIdx.y * 128;
  const int bn = blockIdx.x * 128;
  const int wm = (wave >> 1) * 32;    // 0,32,64,96
  const int wn = (wave & 1) * 64;     // 0,64

  const int lr = tid >> 1;            // staged row 0..127
  const int lc = (tid & 1) * 16;      // 16-half chunk

  v8f acc[2][4] = {};

  const int nslab = K >> 5;
  // prologue: stage slab 0 into buffer 0
  {
    const _Float16* ga = A  + (size_t)(bm + lr) * K + lc;
    const _Float16* gb = Bm + (size_t)(bn + lr) * K + lc;
    async_ld_b128(&As[0][lr * 32 + lc],     ga);
    async_ld_b128(&As[0][lr * 32 + lc + 8], ga + 8);
    async_ld_b128(&Bs[0][lr * 32 + lc],     gb);
    async_ld_b128(&Bs[0][lr * 32 + lc + 8], gb + 8);
  }

  for (int i = 0; i < nslab; ++i) {
    const int cur = i & 1;
    if (i + 1 < nslab) {
      const int k0 = (i + 1) << 5;
      const _Float16* ga = A  + (size_t)(bm + lr) * K + k0 + lc;
      const _Float16* gb = Bm + (size_t)(bn + lr) * K + k0 + lc;
      async_ld_b128(&As[cur ^ 1][lr * 32 + lc],     ga);
      async_ld_b128(&As[cur ^ 1][lr * 32 + lc + 8], ga + 8);
      async_ld_b128(&Bs[cur ^ 1][lr * 32 + lc],     gb);
      async_ld_b128(&Bs[cur ^ 1][lr * 32 + lc + 8], gb + 8);
      wait_async_le4();               // slab i complete (in-order), slab i+1 in flight
    } else {
      wait_async_0();
    }
    __syncthreads();                  // publish slab i to all waves

    v16h a0 = load_frag_a(&As[cur][(wm +  0) * 32], 32, lane);
    v16h a1 = load_frag_a(&As[cur][(wm + 16) * 32], 32, lane);
#pragma unroll
    for (int tn = 0; tn < 4; ++tn) {
      v16h b = load_frag_b(&Bs[cur][(wn + tn * 16) * 32], 32, lane);
      acc[0][tn] = WMMA_F16(a0, b, acc[0][tn]);
      acc[1][tn] = WMMA_F16(a1, b, acc[1][tn]);
    }
    __syncthreads();                  // reads done before buffer reuse
  }

  const int rsel = (lane >> 4) * 8;   // C layout: M = v + 8*(lane>=16)
  const int col  = lane & 15;
#pragma unroll
  for (int tm = 0; tm < 2; ++tm)
#pragma unroll
    for (int tn = 0; tn < 4; ++tn)
#pragma unroll
      for (int v = 0; v < 8; ++v) {
        int row = bm + wm + tm * 16 + v + rsel;
        int cc  = bn + wn + tn * 16 + col;
        C[(size_t)row * N + cc] = (OutT)acc[tm][tn][v];
      }
}

// ---------------------------------------------------------------------------
// Flash attention: one block = (batch b, head h, 64 query rows).
// 4 waves x 16 q-rows each. KV consumed in 64-chunks; Q/K staged via
// async-to-LDS, V transposed through VGPRs into LDS.
// ---------------------------------------------------------------------------
__global__ __launch_bounds__(128)
void attn_flash_wmma(const _Float16* __restrict__ Q,   // [B*Tq, 1024]
                     const _Float16* __restrict__ Kg,  // [B*Tkv, 256]
                     const _Float16* __restrict__ Vg,  // [B*Tkv, 256]
                     _Float16* __restrict__ O) {       // [B*Tq, 1024]
  __shared__ __align__(16) _Float16 Qs[64 * 64];
  __shared__ __align__(16) _Float16 Ks[64 * 64];
  __shared__ __align__(16) _Float16 Vt[64 * 64];          // transposed: [d][kvpos]
  __shared__ __align__(16) _Float16 Ps[4][16 * 64];       // per-wave P tile

  const int tid  = threadIdx.x;
  const int lane = tid & 31;
  const int wave = tid >> 5;
  const int qt = blockIdx.x;     // 0..31
  const int h  = blockIdx.y;     // 0..15
  const int b  = blockIdx.z;     // 0..1
  const int kvh = h >> 2;        // repeat_interleave: head h -> kv head h/4

  const int lr = tid >> 1;             // staging row 0..63
  const int lc = (tid & 1) * 32;       // staging 32-half chunk

  // ---- load Q tile (64 x 64) once, async ----
  const _Float16* Qbase = Q + ((size_t)(b * TQ + qt * 64)) * D_MODEL + h * HEAD_DIM;
#pragma unroll
  for (int i = 0; i < 32; i += 8)
    async_ld_b128(Qs + lr * 64 + lc + i, Qbase + (size_t)lr * D_MODEL + lc + i);
  wait_async_0();
  __syncthreads();

  const int qrow = wave * 16;
  const v16h aq0 = load_frag_a(Qs + qrow * 64 +  0, 64, lane);
  const v16h aq1 = load_frag_a(Qs + qrow * 64 + 32, 64, lane);

  v8f   o[4] = {};
  float mrow[8], lrow[8];
#pragma unroll
  for (int v = 0; v < 8; ++v) { mrow[v] = -3.0e38f; lrow[v] = 0.0f; }

  const _Float16* Kbase = Kg + (size_t)(b * TKV) * KV_DIM + kvh * HEAD_DIM;
  const _Float16* Vbase = Vg + (size_t)(b * TKV) * KV_DIM + kvh * HEAD_DIM;
  const float scale = 0.125f;   // 1/sqrt(64)

  for (int t0 = 0; t0 < TKV; t0 += 64) {
    // ---- stage K async (row-major) and V (transposed, VGPR path) ----
#pragma unroll
    for (int i = 0; i < 32; i += 8)
      async_ld_b128(Ks + lr * 64 + lc + i,
                    Kbase + (size_t)(t0 + lr) * KV_DIM + lc + i);
#pragma unroll
    for (int i = 0; i < 32; i += 8) {
      v8h vv = *(const v8h*)(Vbase + (size_t)(t0 + lr) * KV_DIM + lc + i);
#pragma unroll
      for (int j = 0; j < 8; ++j) Vt[(lc + i + j) * 64 + lr] = vv[j];
    }
    wait_async_0();
    __syncthreads();

    // ---- S = Q * K^T  (16 x 64 per wave) ----
    v8f s[4] = {};
#pragma unroll
    for (int nt = 0; nt < 4; ++nt) {
      v16h bk0 = load_frag_b(Ks + nt * 16 * 64 +  0, 64, lane);
      v16h bk1 = load_frag_b(Ks + nt * 16 * 64 + 32, 64, lane);
      s[nt] = WMMA_F16(aq0, bk0, s[nt]);
      s[nt] = WMMA_F16(aq1, bk1, s[nt]);
    }

    // ---- online softmax (per C-layout row v + 8*(lane>=16)) ----
#pragma unroll
    for (int v = 0; v < 8; ++v) {
      float sv[4];
#pragma unroll
      for (int nt = 0; nt < 4; ++nt) sv[nt] = s[nt][v] * scale;
      float mx = fmaxf(fmaxf(sv[0], sv[1]), fmaxf(sv[2], sv[3]));
#pragma unroll
      for (int m = 1; m < 16; m <<= 1) mx = fmaxf(mx, __shfl_xor(mx, m, 32));
      float mnew  = fmaxf(mrow[v], mx);
      float alpha = __expf(mrow[v] - mnew);
      float rsum = 0.0f;
      float p[4];
#pragma unroll
      for (int nt = 0; nt < 4; ++nt) { p[nt] = __expf(sv[nt] - mnew); rsum += p[nt]; }
#pragma unroll
      for (int m = 1; m < 16; m <<= 1) rsum += __shfl_xor(rsum, m, 32);
      lrow[v] = lrow[v] * alpha + rsum;
      mrow[v] = mnew;
#pragma unroll
      for (int nt = 0; nt < 4; ++nt) o[nt][v] *= alpha;
      const int prow = v + ((lane >> 4) << 3);
#pragma unroll
      for (int nt = 0; nt < 4; ++nt)
        Ps[wave][prow * 64 + nt * 16 + (lane & 15)] = (_Float16)p[nt];
    }

    // ---- O += P * V ---- (P: A-frag from wave-private LDS; V: transposed tile)
    v16h ap0 = load_frag_a(&Ps[wave][0],  64, lane);   // kv 0..31
    v16h ap1 = load_frag_a(&Ps[wave][32], 64, lane);   // kv 32..63
#pragma unroll
    for (int nt = 0; nt < 4; ++nt) {
      v16h bv0 = load_frag_b(Vt + nt * 16 * 64 +  0, 64, lane);
      v16h bv1 = load_frag_b(Vt + nt * 16 * 64 + 32, 64, lane);
      o[nt] = WMMA_F16(ap0, bv0, o[nt]);
      o[nt] = WMMA_F16(ap1, bv1, o[nt]);
    }
    __syncthreads();
  }

  // ---- epilogue: O /= l, write f16 [B*Tq, 1024] ----
  const int rsel = (lane >> 4) << 3;
  const int col  = lane & 15;
#pragma unroll
  for (int nt = 0; nt < 4; ++nt)
#pragma unroll
    for (int v = 0; v < 8; ++v) {
      int row = qt * 64 + qrow + v + rsel;
      float val = o[nt][v] / lrow[v];
      O[((size_t)(b * TQ) + row) * D_MODEL + h * HEAD_DIM + nt * 16 + col] = (_Float16)val;
    }
}

// ---------------------------------------------------------------------------
// Launch
// ---------------------------------------------------------------------------
extern "C" void kernel_launch(void* const* d_in, const int* in_sizes, int n_in,
                              void* d_out, int out_size, void* d_ws, size_t ws_size,
                              hipStream_t stream) {
  const float* query   = (const float*)d_in[0];   // [2,2048,1024]
  const float* context = (const float*)d_in[1];   // [2,2048,1024]
  const float* Wq      = (const float*)d_in[2];   // [1024,1024]
  const float* Wk      = (const float*)d_in[3];   // [256,1024]
  const float* Wv      = (const float*)d_in[4];   // [256,1024]
  const float* Wo      = (const float*)d_in[5];   // [1024,1024]
  float* out = (float*)d_out;                     // [2,2048,1024]

  const size_t M  = (size_t)BATCH * TQ;           // 4096
  const size_t SA = M * D_MODEL;                  // 4M elements

  _Float16* ws   = (_Float16*)d_ws;
  _Float16* q16  = ws;            ws += SA;
  _Float16* c16  = ws;            ws += SA;
  _Float16* wq16 = ws;            ws += (size_t)D_MODEL * D_MODEL;
  _Float16* wk16 = ws;            ws += (size_t)KV_DIM * D_MODEL;
  _Float16* wv16 = ws;            ws += (size_t)KV_DIM * D_MODEL;
  _Float16* wo16 = ws;            ws += (size_t)D_MODEL * D_MODEL;
  _Float16* Q16  = ws;            ws += SA;
  _Float16* K16  = ws;            ws += M * KV_DIM;
  _Float16* V16  = ws;            ws += M * KV_DIM;
  _Float16* AO16 = ws;            ws += SA;

  // 1) precision conversion
  cvt_f32_to_f16<<<2048, 256, 0, stream>>>(query,   q16,  SA);
  cvt_f32_to_f16<<<2048, 256, 0, stream>>>(context, c16,  SA);
  cvt_f32_to_f16<<<1024, 256, 0, stream>>>(Wq, wq16, (size_t)D_MODEL * D_MODEL);
  cvt_f32_to_f16<<<256,  256, 0, stream>>>(Wk, wk16, (size_t)KV_DIM * D_MODEL);
  cvt_f32_to_f16<<<256,  256, 0, stream>>>(Wv, wv16, (size_t)KV_DIM * D_MODEL);
  cvt_f32_to_f16<<<1024, 256, 0, stream>>>(Wo, wo16, (size_t)D_MODEL * D_MODEL);

  // 2) projections: X @ W^T (NT GEMM), f16 outputs
  gemm_nt_wmma<_Float16><<<dim3(D_MODEL / 128, M / 128), 256, 0, stream>>>(
      q16, wq16, Q16, (int)M, D_MODEL, D_MODEL);
  gemm_nt_wmma<_Float16><<<dim3(KV_DIM / 128, M / 128), 256, 0, stream>>>(
      c16, wk16, K16, (int)M, KV_DIM, D_MODEL);
  gemm_nt_wmma<_Float16><<<dim3(KV_DIM / 128, M / 128), 256, 0, stream>>>(
      c16, wv16, V16, (int)M, KV_DIM, D_MODEL);

  // 3) flash attention
  attn_flash_wmma<<<dim3(TQ / 64, N_HEADS, BATCH), 128, 0, stream>>>(
      Q16, K16, V16, AO16);

  // 4) output projection, f32 output
  gemm_nt_wmma<float><<<dim3(D_MODEL / 128, M / 128), 256, 0, stream>>>(
      AO16, wo16, out, (int)M, D_MODEL, D_MODEL);
}
